// Kuramoto_GAT_59184649338990
// MI455X (gfx1250) — matrix operand order repaired and checked by reading, not compiled, not run on hardware
//
#include <hip/hip_runtime.h>
#include <hip/hip_bf16.h>

#define NNODES 100000
#define NEDGES 1600000
#define NFEAT  128
#define NHID   48
#define HEADS  8
#define NCLASS 40
#define NLAYERS 4
#define NEG_SLOPE 0.2f
#define PI_F 3.14159265358979f

typedef __attribute__((ext_vector_type(16))) __bf16 v16bf;
typedef __attribute__((ext_vector_type(8)))  float  v8f;

// ---------------------------------------------------------------------------
// util: fill u32 pattern (graph-capture-safe init of workspace)
// ---------------------------------------------------------------------------
__global__ void kg_fill_u32(unsigned int* __restrict__ p, unsigned int v, long long n) {
  long long t = (long long)blockIdx.x * blockDim.x + threadIdx.x;
  if (t < n) p[t] = v;
}

// ---------------------------------------------------------------------------
// Fold gat_w with att_src/att_dst:  M[k][h] = sum_c gat_w[k, h*48+c] * att[h,c]
// Afterwards a_s = Y @ Msrc  ([N,48]@[48,8]) — eliminates the [N,384] GEMM.
// ---------------------------------------------------------------------------
__global__ void kg_prep_m(const float* __restrict__ gat_w, const float* __restrict__ att_src,
                          const float* __restrict__ att_dst, float* __restrict__ Msrc,
                          float* __restrict__ Mdst) {
  int t = threadIdx.x;
  if (t >= NHID * HEADS) return;
  int k = t >> 3, h = t & 7;
  const float* g  = gat_w + (size_t)k * (HEADS * NHID) + h * NHID;
  const float* as = att_src + h * NHID;
  const float* ad = att_dst + h * NHID;
  float s = 0.f, d = 0.f;
  for (int c = 0; c < NHID; ++c) { s += g[c] * as[c]; d += g[c] * ad[c]; }
  Msrc[t] = s; Mdst[t] = d;
}

// ---------------------------------------------------------------------------
// Encoder: Y = relu(x @ enc_w^T + b), omega = min(Y, pi).
// One wave computes a 16(M) x 48(N) tile via 3 bf16 WMMA accumulators, K=128
// in 4 steps of 32.  A/B fragment layouts per CDNA5 ISA 7.12.2.
// ---------------------------------------------------------------------------
__global__ void __launch_bounds__(256)
kg_encoder_wmma(const float* __restrict__ x, const float* __restrict__ enc_w,
                const float* __restrict__ enc_b, float* __restrict__ Y,
                float* __restrict__ omega) {
  int wave = (int)((blockIdx.x * blockDim.x + threadIdx.x) >> 5);
  int lane = threadIdx.x & 31;
  int half = lane >> 4;      // which 16-lane half of the wave
  int l16  = lane & 15;
  int m0 = wave * 16;
  if (m0 >= NNODES) return;

  v8f acc[3] = {};
  for (int kk = 0; kk < NFEAT; kk += 32) {
    // A fragment: row m0+l16, K = kk + {half*8+e , 16+half*8+e}
    const float* xr = x + (size_t)(m0 + l16) * NFEAT + kk;
    v16bf a;
#pragma unroll
    for (int e = 0; e < 8; ++e) {
      a[e]     = (__bf16)xr[half * 8 + e];
      a[8 + e] = (__bf16)xr[16 + half * 8 + e];
    }
#pragma unroll
    for (int nt = 0; nt < 3; ++nt) {
      // B fragment: B[k][n] = enc_w[n][k];  n = nt*16+l16,  K = kk + {half*8+e , 16+half*8+e}
      const float* wr = enc_w + (size_t)(nt * 16 + l16) * NFEAT + kk + half * 8;
      v16bf b;
#pragma unroll
      for (int e = 0; e < 16; ++e) b[e] = (__bf16)wr[e < 8 ? e : e + 8];
      acc[nt] = __builtin_amdgcn_wmma_f32_16x16x32_bf16(
          false, a, false, b, (short)0, acc[nt], false, false);
    }
  }
#pragma unroll
  for (int nt = 0; nt < 3; ++nt) {
    int col = nt * 16 + l16;
    float bias = enc_b[col];
#pragma unroll
    for (int r = 0; r < 8; ++r) {
      int row = m0 + half * 8 + r;
      float v = acc[nt][r] + bias;
      v = fmaxf(v, 0.f);
      size_t idx = (size_t)row * NHID + col;
      Y[idx] = v;
      omega[idx] = fminf(v, PI_F);
    }
  }
}

// ---------------------------------------------------------------------------
// Per-node attention logits: a_s[n,h] = Y[n,:] @ Msrc[:,h]  (Msrc in LDS)
// ---------------------------------------------------------------------------
__global__ void kg_attn_node(const float* __restrict__ Y, const float* __restrict__ Msrc,
                             const float* __restrict__ Mdst, float* __restrict__ a_s,
                             float* __restrict__ a_d) {
  __shared__ float Ms[NHID * HEADS], Md[NHID * HEADS];
  for (int i = threadIdx.x; i < NHID * HEADS; i += blockDim.x) { Ms[i] = Msrc[i]; Md[i] = Mdst[i]; }
  __syncthreads();
  int n = blockIdx.x * blockDim.x + threadIdx.x;
  if (n >= NNODES) return;
  const float* yr = Y + (size_t)n * NHID;
  float y[NHID];
#pragma unroll
  for (int k = 0; k < NHID; ++k) y[k] = yr[k];
#pragma unroll
  for (int h = 0; h < HEADS; ++h) {
    float s = 0.f, d = 0.f;
    for (int k = 0; k < NHID; ++k) { s += y[k] * Ms[k * 8 + h]; d += y[k] * Md[k * 8 + h]; }
    a_s[(size_t)n * HEADS + h] = s;
    a_d[(size_t)n * HEADS + h] = d;
  }
}

// float atomic max via signed/unsigned int ordering trick (init = -inf bits)
__device__ __forceinline__ void atomicMaxF32(float* addr, float val) {
  if (val >= 0.f) atomicMax((int*)addr, __float_as_int(val));
  else            atomicMin((unsigned int*)addr, __float_as_uint(val));
}

__device__ __forceinline__ float leaky(float e) { return e > 0.f ? e : NEG_SLOPE * e; }

__global__ void kg_edge_max(const int* __restrict__ ei, const float* __restrict__ a_s,
                            const float* __restrict__ a_d, float* __restrict__ m) {
  long long t = (long long)blockIdx.x * blockDim.x + threadIdx.x;
  if (t >= (long long)NEDGES * HEADS) return;
  int edge = (int)(t >> 3), h = (int)(t & 7);
  int row = ei[edge], col = ei[NEDGES + edge];
  float e = leaky(a_s[(size_t)col * HEADS + h] + a_d[(size_t)row * HEADS + h]);
  atomicMaxF32(&m[(size_t)row * HEADS + h], e);
}

__global__ void kg_edge_denom(const int* __restrict__ ei, const float* __restrict__ a_s,
                              const float* __restrict__ a_d, const float* __restrict__ m,
                              float* __restrict__ denom) {
  long long t = (long long)blockIdx.x * blockDim.x + threadIdx.x;
  if (t >= (long long)NEDGES * HEADS) return;
  int edge = (int)(t >> 3), h = (int)(t & 7);
  int row = ei[edge], col = ei[NEDGES + edge];
  float e = leaky(a_s[(size_t)col * HEADS + h] + a_d[(size_t)row * HEADS + h]);
  float ee = __expf(e - m[(size_t)row * HEADS + h]);
  atomicAdd(&denom[(size_t)row * HEADS + h], ee);
}

__global__ void kg_edge_w(const int* __restrict__ ei, const float* __restrict__ a_s,
                          const float* __restrict__ a_d, const float* __restrict__ m,
                          const float* __restrict__ denom, float* __restrict__ w) {
  int edge = blockIdx.x * blockDim.x + threadIdx.x;
  if (edge >= NEDGES) return;
  int row = ei[edge], col = ei[NEDGES + edge];
  float s = 0.f;
#pragma unroll
  for (int h = 0; h < HEADS; ++h) {
    float e = leaky(a_s[(size_t)col * HEADS + h] + a_d[(size_t)row * HEADS + h]);
    float ee = __expf(e - m[(size_t)row * HEADS + h]);
    s += ee / (denom[(size_t)row * HEADS + h] + 1e-16f);
  }
  w[edge] = s * (1.f / HEADS);
}

// ---------------------------------------------------------------------------
// Kuramoto edge pass: gather Y[col] (L2-resident), compute cos/sin on the fly,
// scatter-add w*(y,cos y,sin y) into agg[row].  192 threads = 4 edges/block.
// ---------------------------------------------------------------------------
__global__ void __launch_bounds__(192)
kg_edge_scatter(const int* __restrict__ ei, const float* __restrict__ Y,
                const float* __restrict__ w, float* __restrict__ agg) {
  int edge = blockIdx.x * 4 + threadIdx.x / NHID;
  int f = threadIdx.x % NHID;
  if (edge >= NEDGES) return;
  int row = ei[edge], col = ei[NEDGES + edge];
  float we = w[edge];
  float y = Y[(size_t)col * NHID + f];
  float s, c;
  __sincosf(y, &s, &c);
  float* ar = agg + (size_t)row * (3 * NHID);
  atomicAdd(ar + f,            we * y);
  atomicAdd(ar + NHID + f,     we * c);
  atomicAdd(ar + 2 * NHID + f, we * s);
}

__global__ void kg_node_update(const float* __restrict__ agg, const float* __restrict__ omega,
                               const float* __restrict__ ks_raw, float* __restrict__ Y) {
  int t = blockIdx.x * blockDim.x + threadIdx.x;
  if (t >= NNODES * NHID) return;
  int n = t / NHID, f = t % NHID;
  float Ks = log1pf(__expf(ks_raw[0]));   // softplus
  const float* ar = agg + (size_t)n * (3 * NHID);
  float phi = ar[f], cr = ar[NHID + f], sr = ar[2 * NHID + f];
  float R = sqrtf(cr * cr + sr * sr + 1e-12f);
  float y = Y[t];
  Y[t] = y + (omega[t] + Ks * R * __sinf(phi - y));   // DT = 1
}

// ---------------------------------------------------------------------------
// Decoder: out = Y @ dec_w^T + dec_b.  M=16/wave, N padded 40->48, K padded
// 48->64 (zeros).  Same WMMA tiling as encoder.
// ---------------------------------------------------------------------------
__global__ void __launch_bounds__(256)
kg_decoder_wmma(const float* __restrict__ Y, const float* __restrict__ dec_w,
                const float* __restrict__ dec_b, float* __restrict__ out) {
  int wave = (int)((blockIdx.x * blockDim.x + threadIdx.x) >> 5);
  int lane = threadIdx.x & 31;
  int half = lane >> 4, l16 = lane & 15;
  int m0 = wave * 16;
  if (m0 >= NNODES) return;

  v8f acc[3] = {};
  const float* yr = Y + (size_t)(m0 + l16) * NHID;
#pragma unroll
  for (int kk = 0; kk < 64; kk += 32) {
    v16bf a;
#pragma unroll
    for (int e = 0; e < 8; ++e) {
      int k0 = kk + half * 8 + e;
      int k1 = kk + 16 + half * 8 + e;
      a[e]     = (__bf16)(k0 < NHID ? yr[k0] : 0.f);
      a[8 + e] = (__bf16)(k1 < NHID ? yr[k1] : 0.f);
    }
#pragma unroll
    for (int nt = 0; nt < 3; ++nt) {
      int n = nt * 16 + l16;
      const float* wr = dec_w + (size_t)n * NHID;
      v16bf b;
#pragma unroll
      for (int e = 0; e < 16; ++e) {
        int k = kk + half * 8 + (e < 8 ? e : e + 8);
        b[e] = (__bf16)((n < NCLASS && k < NHID) ? wr[k] : 0.f);
      }
      acc[nt] = __builtin_amdgcn_wmma_f32_16x16x32_bf16(
          false, a, false, b, (short)0, acc[nt], false, false);
    }
  }
#pragma unroll
  for (int nt = 0; nt < 3; ++nt) {
    int col = nt * 16 + l16;
    if (col >= NCLASS) continue;
    float bias = dec_b[col];
#pragma unroll
    for (int r = 0; r < 8; ++r) {
      int row = m0 + half * 8 + r;
      out[(size_t)row * NCLASS + col] = acc[nt][r] + bias;
    }
  }
}

// ---------------------------------------------------------------------------
extern "C" void kernel_launch(void* const* d_in, const int* in_sizes, int n_in,
                              void* d_out, int out_size, void* d_ws, size_t ws_size,
                              hipStream_t stream) {
  (void)in_sizes; (void)n_in; (void)out_size; (void)ws_size;
  const float* x       = (const float*)d_in[0];
  const int*   ei      = (const int*)d_in[1];
  const float* enc_w   = (const float*)d_in[2];
  const float* enc_b   = (const float*)d_in[3];
  const float* gat_w   = (const float*)d_in[4];
  const float* att_src = (const float*)d_in[5];
  const float* att_dst = (const float*)d_in[6];
  const float* ks_raw  = (const float*)d_in[7];
  const float* dec_w   = (const float*)d_in[8];
  const float* dec_b   = (const float*)d_in[9];
  float* out = (float*)d_out;

  size_t off = 0;
  auto alloc = [&](size_t nfloats) {
    float* p = (float*)((char*)d_ws + off);
    off += nfloats * sizeof(float);
    return p;
  };
  float* Y     = alloc((size_t)NNODES * NHID);      // 19.2 MB
  float* omega = alloc((size_t)NNODES * NHID);      // 19.2 MB
  float* a_s   = alloc((size_t)NNODES * HEADS);
  float* a_d   = alloc((size_t)NNODES * HEADS);
  float* m     = alloc((size_t)NNODES * HEADS);
  float* denom = alloc((size_t)NNODES * HEADS);
  float* Msrc  = alloc(NHID * HEADS);
  float* Mdst  = alloc(NHID * HEADS);
  float* w     = alloc(NEDGES);
  float* agg   = alloc((size_t)NNODES * 3 * NHID);  // 57.6 MB

  const int waves = NNODES / 16;                    // 6250
  const int gemmBlocks = (waves * 32 + 255) / 256;

  kg_prep_m<<<1, NHID * HEADS, 0, stream>>>(gat_w, att_src, att_dst, Msrc, Mdst);
  kg_encoder_wmma<<<gemmBlocks, 256, 0, stream>>>(x, enc_w, enc_b, Y, omega);
  kg_attn_node<<<(NNODES + 255) / 256, 256, 0, stream>>>(Y, Msrc, Mdst, a_s, a_d);

  long long nh = (long long)NNODES * HEADS;
  kg_fill_u32<<<(unsigned)((nh + 255) / 256), 256, 0, stream>>>((unsigned*)m, 0xFF800000u, nh);     // -inf
  kg_fill_u32<<<(unsigned)((nh + 255) / 256), 256, 0, stream>>>((unsigned*)denom, 0u, nh);

  long long eh = (long long)NEDGES * HEADS;
  kg_edge_max  <<<(unsigned)((eh + 255) / 256), 256, 0, stream>>>(ei, a_s, a_d, m);
  kg_edge_denom<<<(unsigned)((eh + 255) / 256), 256, 0, stream>>>(ei, a_s, a_d, m, denom);
  kg_edge_w    <<<(NEDGES + 255) / 256, 256, 0, stream>>>(ei, a_s, a_d, m, denom, w);

  long long aggn = (long long)NNODES * 3 * NHID;
  for (int l = 0; l < NLAYERS; ++l) {
    kg_fill_u32<<<(unsigned)((aggn + 255) / 256), 256, 0, stream>>>((unsigned*)agg, 0u, aggn);
    kg_edge_scatter<<<(NEDGES + 3) / 4, 192, 0, stream>>>(ei, Y, w, agg);
    kg_node_update<<<(NNODES * NHID + 255) / 256, 256, 0, stream>>>(agg, omega, ks_raw, Y);
  }

  kg_decoder_wmma<<<gemmBlocks, 256, 0, stream>>>(Y, dec_w, dec_b, out);
}
